// UniPhyBlock_86268713108041
// MI455X (gfx1250) — compile-verified, matrix operand order checked by reading, not run on hardware
//
#include <hip/hip_runtime.h>
#include <hip/hip_bf16.h>
#include <math.h>

// ---------------------------------------------------------------------------
// UniPhyBlock for MI455X (gfx1250, wave32):
//  - all GEMMs via v_wmma_f32_16x16x32_bf16 (f32 accum)
//  - weight tiles staged global->LDS by the Tensor Data Mover (TDM) with
//    LDS row padding encoded in the D# descriptor (CDNA5 ISA)
//  - double-buffered B-fragment pipeline so ds_load_b128 overlaps WMMA
// ---------------------------------------------------------------------------

typedef __bf16 bf16;
typedef __attribute__((ext_vector_type(16))) __bf16 v16bf;
typedef __attribute__((ext_vector_type(8)))  __bf16 v8bf;
typedef __attribute__((ext_vector_type(4)))  __bf16 v4bf;
typedef __attribute__((ext_vector_type(8)))  float  v8f;
typedef __attribute__((ext_vector_type(4)))  unsigned int u32x4;
typedef __attribute__((ext_vector_type(8)))  int i32x8;
typedef __attribute__((ext_vector_type(4)))  int i32x4;

#define DEVINL __device__ __forceinline__

constexpr int kB = 2, kT = 16, kD = 64;
constexpr int kC = 128;
constexpr int kN = kB * kT;      // 32 images (B*T)
constexpr int kP = 64 * 64;      // 4096 pixels / image
constexpr int kNP = kN * kP;     // 131072 pixel-rows

DEVINL v8f wmma_bf16(v16bf a, v16bf b, v8f c) {
  return __builtin_amdgcn_wmma_f32_16x16x32_bf16(false, a, false, b,
                                                 (short)0, c, false, false);
}

// 16x32 bf16 fragment loader (CDNA5 ISA 7.12.2).  A tiles are row-major MxK,
// B tiles are stored transposed (NxK) so the same mapping applies: lane L
// holds row (L&15); half i=0..7 -> K = koff+i, i=8..15 -> K = 8+koff+i with
// koff = 8*(L>=16).  Both halves are contiguous -> two ds_load_b128.
DEVINL v16bf load_frag(const bf16* s, int ld) {
  int lane = threadIdx.x & 31;
  const bf16* p = s + (lane & 15) * ld + ((lane >> 4) << 3);
  v8bf lo = *(const v8bf*)p;
  v8bf hi = *(const v8bf*)(p + 16);
  return __builtin_shufflevector(lo, hi, 0,1,2,3,4,5,6,7,8,9,10,11,12,13,14,15);
}

// Flat LDS address -> LDS byte offset (ISA 10.2: low 32 address bits).
DEVINL unsigned lds_off(const void* p) { return (unsigned)(uintptr_t)p; }

// TDM: load `rows` x `cols`(bf16) contiguous 2D tile from global into LDS,
// appending pad DWORDs after every row so the LDS row stride stays
// bank-conflict friendly.  padIntervalCode: pad after 2<<code DWORDs;
// padAmountCode: (code+1) DWORDs of padding.  D# per CDNA5 ISA.
// Must be issued by a single wave (TDM ignores EXEC); follow with
// s_wait_tensorcnt(0) on that wave + a workgroup barrier.
DEVINL void tdm_load_rows(const bf16* gsrc, unsigned ldsByteAddr,
                          int rows, int cols,
                          int padIntervalCode, int padAmountCode) {
  unsigned long long ga = (unsigned long long)(uintptr_t)gsrc;
  u32x4 g0;
  g0[0] = 1u;                                             // count=1 (valid)
  g0[1] = ldsByteAddr;                                    // lds_addr[31:0]
  g0[2] = (unsigned)(ga & 0xffffffffu);                   // global_addr lo
  g0[3] = (unsigned)((ga >> 32) & 0x01ffffffu) | (2u << 30);  // hi | type=2
  i32x8 g1;
  g1[0] = (1 << 16)                                       // data_size = 2B
        | (1 << 20)                                       // pad_enable
        | (padIntervalCode << 22) | (padAmountCode << 25);
  g1[1] = (cols & 0xffff) << 16;                          // tensor_dim0 lo16
  g1[2] = ((cols >> 16) & 0xffff) | ((rows & 0xffff) << 16); // dim0 hi|dim1 lo
  g1[3] = ((rows >> 16) & 0xffff) | ((cols & 0xffff) << 16); // dim1 hi|tile0
  g1[4] = rows & 0xffff;                                  // tile_dim1 (tile2=0)
  g1[5] = cols;                                           // dim0_stride lo32
  g1[6] = 0;                                              // stride hi bits
  g1[7] = 0;
  i32x4 z4 = {0, 0, 0, 0};
  i32x8 z8 = {0, 0, 0, 0, 0, 0, 0, 0};
  __builtin_amdgcn_tensor_load_to_lds(g0, g1, z4, z4, z8, 0);
}

// --------------------------- weight preparation ----------------------------
// All GEMM B-matrices converted to bf16 and transposed to [N][K]:
//  wtapT[tap][co][ci]  (conv),  wE/wF[j][k] = [[Er,Ei],[-Ei,Er]]^T (complex),
//  w1T[e][m][ci], w2T[e][c][m] (MoE).
__global__ void k_prep(const float* __restrict__ conv_w,
                       const float* __restrict__ Er, const float* __restrict__ Ei,
                       const float* __restrict__ Fr, const float* __restrict__ Fi,
                       const float* __restrict__ ew1, const float* __restrict__ ew2,
                       bf16* wtap, bf16* wE, bf16* wF, bf16* w1b, bf16* w2b) {
  int i = blockIdx.x * 256 + threadIdx.x;
  if (i < 9 * 128 * 128) {               // conv taps, transposed to [co][ci]
    int tap = i >> 14, rem = i & 16383, co = rem >> 7, ci = rem & 127;
    wtap[i] = (bf16)conv_w[(co * 128 + ci) * 9 + tap];
  }
  if (i < 16384) {                       // complex mixers, [j][k]
    int j = i >> 7, k = i & 127;
    float ve, vf;
    if (k < 64) {
      ve = (j < 64) ? Er[k * 64 + j] : Ei[k * 64 + (j - 64)];
      vf = (j < 64) ? Fr[k * 64 + j] : Fi[k * 64 + (j - 64)];
    } else {
      int kk = k - 64;
      ve = (j < 64) ? -Ei[kk * 64 + j] : Er[kk * 64 + (j - 64)];
      vf = (j < 64) ? -Fi[kk * 64 + j] : Fr[kk * 64 + (j - 64)];
    }
    wE[i] = (bf16)ve; wF[i] = (bf16)vf;
  }
  if (i < 4 * 128 * 256) {               // e_w1 (E,C,M) -> [e][m][ci]
    int e = i >> 15, rem = i & 32767, m = rem >> 7, ci = rem & 127;
    w1b[i] = (bf16)ew1[(e * 128 + ci) * 256 + m];
  }
  if (i < 4 * 256 * 128) {               // e_w2 (E,M,C) -> [e][c][m]
    int e = i >> 15, rem = i & 32767, c = rem >> 8, m = rem & 255;
    w2b[i] = (bf16)ew2[(e * 256 + m) * 128 + c];
  }
}

// ----------------- LN1: per-pixel LN over C=128 + transpose ----------------
__global__ void k_ln1(const float* __restrict__ xre, const float* __restrict__ xim,
                      const float* __restrict__ g, const float* __restrict__ b,
                      bf16* __restrict__ xn, float* __restrict__ xflat) {
  __shared__ float tile[128][65];
  __shared__ float mu[64], rs[64];
  int p0 = blockIdx.x * 64;
  int n = p0 >> 12, pl = p0 & 4095;
  int t = threadIdx.x, pi = t & 63, c0 = t >> 6;
  for (int c = c0; c < 128; c += 4) {
    float v = (c < 64) ? xre[((size_t)(n * 64 + c)) * 4096 + pl + pi]
                       : xim[((size_t)(n * 64 + (c - 64))) * 4096 + pl + pi];
    tile[c][pi] = v;
  }
  __syncthreads();
  if (t < 64) {
    float s = 0.f;
    for (int c = 0; c < 128; ++c) s += tile[c][t];
    float m = s * (1.0f / 128.0f), v = 0.f;
    for (int c = 0; c < 128; ++c) { float d = tile[c][t] - m; v += d * d; }
    mu[t] = m; rs[t] = rsqrtf(v * (1.0f / 128.0f) + 1e-5f);
  }
  __syncthreads();
  int c = t & 127, pg0 = t >> 7;
  float gg = g[c], bb = b[c];
  for (int p = pg0; p < 64; p += 2) {
    float xv = tile[c][p];
    size_t gi = (size_t)(p0 + p) * 128 + c;
    xn[gi] = (bf16)((xv - mu[p]) * rs[p] * gg + bb);
    xflat[gi] = xv;
  }
}

// ------------- conv 3x3 as 9-tap implicit GEMM + bias + residual -----------
// Block: 128 pixels x 128 out-channels.  8 waves, each 16 pixels x 128 cols.
// Weight tile of each tap is DMA'd by the TDM while threads gather the
// shifted activation tile.
__global__ void k_conv(const bf16* __restrict__ xn, const bf16* __restrict__ wtap,
                       const float* __restrict__ conv_b,
                       const float* __restrict__ xflat, float* __restrict__ xpost) {
  extern __shared__ char smem[];
  bf16* As = (bf16*)smem;                  // 128 x 136 (pixels x K)
  bf16* Bs = (bf16*)(smem + 128 * 136 * 2);// 128 x 136 (N x K, transposed W)
  const int LD = 136;
  int tid = threadIdx.x, lane = tid & 31, wid = tid >> 5;
  int p0 = blockIdx.x * 128, n = p0 >> 12, pl0 = p0 & 4095;

  v8f acc[8] = {};
  for (int tap = 0; tap < 9; ++tap) {
    int dy = tap / 3 - 1, dx = tap % 3 - 1;
    __syncthreads();                       // all readers of As/Bs done
    if (wid == 0)                          // TDM: 128x128 bf16 tile, pad 4 DW
      tdm_load_rows(wtap + (size_t)tap * 16384, lds_off(Bs), 128, 128, 5, 3);
    if (tap < 8) __builtin_prefetch(wtap + (size_t)(tap + 1) * 16384, 0, 0);
    for (int c = tid; c < 2048; c += 256) {
      int row = c >> 4, col = (c & 15) << 3;
      int p = pl0 + row, hh = (p >> 6) + dy, ww = (p & 63) + dx;
      uint4 v = {0, 0, 0, 0};
      if (hh >= 0 && hh < 64 && ww >= 0 && ww < 64)
        v = *(const uint4*)&xn[((size_t)(n * 4096 + hh * 64 + ww)) * 128 + col];
      *(uint4*)&As[row * LD + col] = v;
    }
    if (wid == 0) __builtin_amdgcn_s_wait_tensorcnt(0);
    __syncthreads();
#pragma unroll
    for (int kc = 0; kc < 4; ++kc) {
      v16bf a = load_frag(&As[(wid * 16) * LD + kc * 32], LD);
      v16bf bcur = load_frag(&Bs[kc * 32], LD);
#pragma unroll
      for (int nt = 0; nt < 8; ++nt) {
        v16bf bnxt;
        if (nt < 7) bnxt = load_frag(&Bs[((nt + 1) * 16) * LD + kc * 32], LD);
        acc[nt] = wmma_bf16(a, bcur, acc[nt]);
        if (nt < 7) bcur = bnxt;
      }
    }
  }
#pragma unroll
  for (int nt = 0; nt < 8; ++nt)
#pragma unroll
    for (int r = 0; r < 8; ++r) {
      int row = wid * 16 + r + ((lane >> 4) << 3);
      int col = nt * 16 + (lane & 15);
      size_t gi = (size_t)(p0 + row) * 128 + col;
      xpost[gi] = acc[nt][r] + conv_b[col] + xflat[gi];
    }
}

// ------------ generic (NPx128) x (128x128) GEMM (E / F projections) --------
__global__ void k_gemm128(const float* __restrict__ A, const bf16* __restrict__ Bg,
                          float* __restrict__ Cout) {
  extern __shared__ char smem[];
  bf16* As = (bf16*)smem;
  bf16* Bs = (bf16*)(smem + 128 * 136 * 2);
  const int LD = 136;
  int tid = threadIdx.x, lane = tid & 31, wid = tid >> 5;
  size_t p0 = (size_t)blockIdx.x * 128;
  if (wid == 0) tdm_load_rows(Bg, lds_off(Bs), 128, 128, 5, 3);
  for (int c = tid; c < 4096; c += 256) {
    int row = c >> 5, col = (c & 31) << 2;
    float4 v = *(const float4*)&A[(p0 + row) * 128 + col];
    bf16* d = &As[row * LD + col];
    d[0] = (bf16)v.x; d[1] = (bf16)v.y; d[2] = (bf16)v.z; d[3] = (bf16)v.w;
  }
  if (wid == 0) __builtin_amdgcn_s_wait_tensorcnt(0);
  __syncthreads();
  v8f acc[8] = {};
#pragma unroll
  for (int kc = 0; kc < 4; ++kc) {
    v16bf a = load_frag(&As[(wid * 16) * LD + kc * 32], LD);
    v16bf bcur = load_frag(&Bs[kc * 32], LD);
#pragma unroll
    for (int nt = 0; nt < 8; ++nt) {
      v16bf bnxt;
      if (nt < 7) bnxt = load_frag(&Bs[((nt + 1) * 16) * LD + kc * 32], LD);
      acc[nt] = wmma_bf16(a, bcur, acc[nt]);
      if (nt < 7) bcur = bnxt;
    }
  }
#pragma unroll
  for (int nt = 0; nt < 8; ++nt)
#pragma unroll
    for (int r = 0; r < 8; ++r) {
      int row = wid * 16 + r + ((lane >> 4) << 3);
      int col = nt * 16 + (lane & 15);
      Cout[(p0 + row) * 128 + col] = acc[nt][r];
    }
}

// ------------------------- spatial mean over H,W ---------------------------
__global__ void k_mean(const float* __restrict__ xeig, float* __restrict__ xmean) {
  __shared__ float sd[256];
  int n = blockIdx.x, t = threadIdx.x, c = t & 127, pg = t >> 7;
  float s = 0.f;
  for (int p = pg; p < 4096; p += 2)
    s += xeig[((size_t)n * 4096 + p) * 128 + c];
  sd[t] = s;
  __syncthreads();
  if (pg == 0) xmean[n * 128 + c] = (sd[c] + sd[128 + c]) * (1.0f / 4096.0f);
}

// ------------------------------ mix matvec --------------------------------
__global__ void k_mix(const float* __restrict__ xmean, const float* __restrict__ mix_w,
                      const float* __restrict__ mix_b, float* __restrict__ fcat) {
  int i = blockIdx.x * 256 + threadIdx.x;  // < 4096
  int n = i >> 7, co = i & 127;
  float s = mix_b[co];
  for (int ci = 0; ci < 128; ++ci) s += xmean[n * 128 + ci] * mix_w[ci * 128 + co];
  fcat[i] = s;
}

// ----------------------- flux scan over T (sequential) ---------------------
__global__ void k_fluxscan(const float* __restrict__ fcat, const float* __restrict__ nu,
                           const float* __restrict__ theta,
                           const float* __restrict__ flux_re,
                           const float* __restrict__ flux_im,
                           float* __restrict__ fluxseq) {
  int t = threadIdx.x; if (t >= 128) return;
  int b = t >> 6, d = t & 63;
  float x = nu[d];
  float sp = (x > 20.f) ? x : log1pf(expf(x));
  float mag = expf(-sp);
  float ar = mag * cosf(theta[d]), ai = mag * sinf(theta[d]);
  float fr0 = flux_re[b * 64 + d], fi0 = flux_im[b * 64 + d];
  float accr = 0.f, acci = 0.f, cr = 1.f, ci = 0.f;
  for (int tt = 0; tt < 16; ++tt) {
    int n = b * 16 + tt;
    float xr = fcat[n * 128 + d], xi = fcat[n * 128 + 64 + d];
    float nr = ar * accr - ai * acci + xr;
    float ni = ar * acci + ai * accr + xi;
    accr = nr; acci = ni;
    float ncr = cr * ar - ci * ai, nci = cr * ai + ci * ar;
    cr = ncr; ci = nci;
    fluxseq[n * 128 + d]      = accr + fr0 * cr - fi0 * ci;
    fluxseq[n * 128 + 64 + d] = acci + fr0 * ci + fi0 * cr;
  }
}

// ---------------- source / gate / op_decay / op_forcing --------------------
__global__ void k_srcgate(const float* __restrict__ ocat, const float* __restrict__ src_w,
                          const float* __restrict__ src_b, const float* __restrict__ gate_w,
                          const float* __restrict__ gate_b, const float* __restrict__ lam_a,
                          const float* __restrict__ lam_b, const float* __restrict__ dt,
                          float* __restrict__ source, float* __restrict__ gate,
                          float* __restrict__ opdec, float* __restrict__ opf) {
  int i = blockIdx.x * 256 + threadIdx.x;  // < 4096
  int n = i >> 7, co = i & 127;
  float s = src_b[co];
  for (int ci = 0; ci < 128; ++ci) s += ocat[n * 128 + ci] * src_w[ci * 128 + co];
  source[i] = s;
  if (co < 64) {
    int d = co;
    float gs = gate_b[d];
    for (int ci = 0; ci < 128; ++ci) gs += ocat[n * 128 + ci] * gate_w[ci * 64 + d];
    gate[n * 64 + d] = 1.f / (1.f + expf(-gs));
    int tt = n & 15;
    float la = lam_a[d];
    float sp = (la > 20.f) ? la : log1pf(expf(la));
    float lr = -sp, li = lam_b[d];
    float dtv = dt[tt];                    // DT_REF = 1
    float m = expf(lr * dtv);
    float odr = m * cosf(li * dtv), odi = m * sinf(li * dtv);
    opdec[n * 128 + d] = odr; opdec[n * 128 + 64 + d] = odi;
    float numr = odr - 1.f, numi = odi;
    float den = lr * lr + li * li;
    opf[n * 128 + d]      = (numr * lr + numi * li) / den;
    opf[n * 128 + 64 + d] = (numi * lr - numr * li) / den;
  }
}

// -------------------- forcing + op_forcing + h0 seed -> u_t ----------------
__global__ void k_forcing(const float* __restrict__ xeig, const float* __restrict__ gate,
                          const float* __restrict__ source, const float* __restrict__ opf,
                          const float* __restrict__ opdec, const float* __restrict__ h_re,
                          const float* __restrict__ h_im, float* __restrict__ u) {
  size_t i = (size_t)blockIdx.x * 256 + threadIdx.x;   // < NP*64
  int d = (int)(i & 63);
  size_t pg = i >> 6;
  int n = (int)(pg >> 12);
  float xr = xeig[pg * 128 + d], xi = xeig[pg * 128 + 64 + d];
  float g = gate[n * 64 + d];
  float sr = source[n * 128 + d], si = source[n * 128 + 64 + d];
  float fr = xr * g + sr * (1.f - g);
  float fi = xi * g + si * (1.f - g);
  float ofr = opf[n * 128 + d], ofi = opf[n * 128 + 64 + d];
  float ur = fr * ofr - fi * ofi;
  float ui = fr * ofi + fi * ofr;
  if ((n & 15) == 0) {
    int b = n >> 4;
    int pimg = (int)(pg & 4095);
    float hr = h_re[((size_t)(b * 4096 + pimg)) * 64 + d];
    float hi = h_im[((size_t)(b * 4096 + pimg)) * 64 + d];
    float odr = opdec[n * 128 + d], odi = opdec[n * 128 + 64 + d];
    ur += hr * odr - hi * odi;
    ui += hr * odi + hi * odr;
  }
  u[pg * 128 + d] = ur; u[pg * 128 + 64 + d] = ui;
}

// ------------------- state recurrence over T (in place) --------------------
__global__ void k_scanT(const float* __restrict__ opdec, float* __restrict__ u) {
  size_t i = (size_t)blockIdx.x * 256 + threadIdx.x;   // < B*P*64
  int d = (int)(i & 63);
  size_t r = i >> 6;
  int pimg = (int)(r & 4095), b = (int)(r >> 12);
  float accr = 0.f, acci = 0.f;
  for (int tt = 0; tt < 16; ++tt) {
    int n = b * 16 + tt;
    size_t base = ((size_t)n * 4096 + pimg) * 128;
    float odr = opdec[n * 128 + d], odi = opdec[n * 128 + 64 + d];
    float ur = u[base + d], ui = u[base + 64 + d];
    float nr = odr * accr - odi * acci + ur;
    float ni = odr * acci + odi * accr + ui;
    accr = nr; acci = ni;
    u[base + d] = accr; u[base + 64 + d] = acci;
  }
}

// -------------------- LN2 (wave per pixel, shfl reduce) --------------------
__global__ void k_ln2(const float* __restrict__ xdec, const float* __restrict__ g,
                      const float* __restrict__ b, float* __restrict__ onf,
                      bf16* __restrict__ onb) {
  int lane = threadIdx.x & 31, wid = threadIdx.x >> 5;
  size_t pix = (size_t)blockIdx.x * 8 + wid;
  float4 v = *(const float4*)&xdec[pix * 128 + lane * 4];
  float s = v.x + v.y + v.z + v.w;
#pragma unroll
  for (int m = 16; m >= 1; m >>= 1) s += __shfl_xor(s, m, 32);
  float mu = s * (1.0f / 128.0f);
  float dx = v.x - mu, dy = v.y - mu, dz = v.z - mu, dw = v.w - mu;
  float q = dx * dx + dy * dy + dz * dz + dw * dw;
#pragma unroll
  for (int m = 16; m >= 1; m >>= 1) q += __shfl_xor(q, m, 32);
  float rs = rsqrtf(q * (1.0f / 128.0f) + 1e-5f);
  int c = lane * 4;
  float4 gg = *(const float4*)&g[c];
  float4 bb = *(const float4*)&b[c];
  float y0 = dx * rs * gg.x + bb.x, y1 = dy * rs * gg.y + bb.y;
  float y2 = dz * rs * gg.z + bb.z, y3 = dw * rs * gg.w + bb.w;
  float4 yo = {y0, y1, y2, y3};
  *(float4*)&onf[pix * 128 + c] = yo;
  v4bf o; o[0] = (bf16)y0; o[1] = (bf16)y1; o[2] = (bf16)y2; o[3] = (bf16)y3;
  *(v4bf*)&onb[pix * 128 + c] = o;
}

// ----- fused MoE (router + 2 GEMMs x 4 experts) + residual + transpose -----
// Prob weighting folded into hidden rows: moe = sum_e (diag(p_e)*gelu(H_e))@W2_e
// so WMMA accumulates across experts.  Expert weights staged by the TDM.
__global__ void k_moe(const bf16* __restrict__ onb, const float* __restrict__ onf,
                      const float* __restrict__ xpost, const bf16* __restrict__ w1b,
                      const bf16* __restrict__ w2b, const float* __restrict__ r_w,
                      const float* __restrict__ r_b, const float* __restrict__ e_b1,
                      const float* __restrict__ e_b2, float* __restrict__ out) {
  extern __shared__ char smem[];
  bf16* As = (bf16*)smem;                                  // 64 x 136
  bf16* Hs = (bf16*)(smem + 64 * 136 * 2);                 // 64 x 264
  bf16* Bs = (bf16*)(smem + 64 * 136 * 2 + 64 * 264 * 2);  // <= 69632 B
  float* Pr = (float*)(smem + 64 * 136 * 2 + 64 * 264 * 2 + 69632);  // 64x4
  float* Zs = Pr + 64 * 4;                                 // 64 x 133
  int tid = threadIdx.x, lane = tid & 31, wid = tid >> 5;
  int p0 = blockIdx.x * 64, n = p0 >> 12;

  for (int c = tid; c < 1024; c += 256) {
    int row = c >> 4, col = (c & 15) << 3;
    *(uint4*)&As[row * 136 + col] =
        *(const uint4*)&onb[((size_t)(p0 + row)) * 128 + col];
  }
  __syncthreads();
  if (tid < 64) {  // router softmax
    float l0 = r_b[0], l1 = r_b[1], l2 = r_b[2], l3 = r_b[3];
    for (int ci = 0; ci < 128; ++ci) {
      float a = (float)As[tid * 136 + ci];
      l0 += a * r_w[ci * 4 + 0]; l1 += a * r_w[ci * 4 + 1];
      l2 += a * r_w[ci * 4 + 2]; l3 += a * r_w[ci * 4 + 3];
    }
    float mx = fmaxf(fmaxf(l0, l1), fmaxf(l2, l3));
    float e0 = expf(l0 - mx), e1 = expf(l1 - mx);
    float e2 = expf(l2 - mx), e3 = expf(l3 - mx);
    float inv = 1.f / (e0 + e1 + e2 + e3);
    Pr[tid * 4 + 0] = e0 * inv; Pr[tid * 4 + 1] = e1 * inv;
    Pr[tid * 4 + 2] = e2 * inv; Pr[tid * 4 + 3] = e3 * inv;
  }
  __syncthreads();

  int mi = wid & 3, ni = wid >> 2;   // GEMM2: 16 px x 64 cols per wave
  v8f macc[4] = {};
  for (int e = 0; e < 4; ++e) {
    // GEMM1: B = w1T[e] : 256(N=m) x 128(K=ci), LDS stride 136
    if (wid == 0) {
      tdm_load_rows(w1b + (size_t)e * 32768, lds_off(Bs), 256, 128, 5, 3);
      __builtin_amdgcn_s_wait_tensorcnt(0);
    }
    __syncthreads();
    v8f hacc[8] = {};
#pragma unroll
    for (int kc = 0; kc < 4; ++kc) {
      v16bf a = load_frag(&As[(mi * 16) * 136 + kc * 32], 136);
      v16bf bcur = load_frag(&Bs[(ni * 128) * 136 + kc * 32], 136);
#pragma unroll
      for (int j = 0; j < 8; ++j) {
        v16bf bnxt;
        if (j < 7) bnxt = load_frag(&Bs[(ni * 128 + (j + 1) * 16) * 136 + kc * 32], 136);
        hacc[j] = wmma_bf16(a, bcur, hacc[j]);
        if (j < 7) bcur = bnxt;
      }
    }
#pragma unroll
    for (int j = 0; j < 8; ++j)
#pragma unroll
      for (int r = 0; r < 8; ++r) {
        int row = mi * 16 + r + ((lane >> 4) << 3);
        int col = ni * 128 + j * 16 + (lane & 15);
        float v = hacc[j][r] + e_b1[e * 256 + col];
        float gl = 0.5f * v * (1.f + tanhf(0.7978845608f * (v + 0.044715f * v * v * v)));
        Hs[row * 264 + col] = (bf16)(gl * Pr[row * 4 + e]);
      }
    __syncthreads();
    // GEMM2: B = w2T[e] : 128(N=c) x 256(K=m), LDS stride 264
    if (wid == 0) {
      tdm_load_rows(w2b + (size_t)e * 32768, lds_off(Bs), 128, 256, 6, 3);
      __builtin_amdgcn_s_wait_tensorcnt(0);
    }
    __syncthreads();
#pragma unroll
    for (int kc = 0; kc < 8; ++kc) {
      v16bf a = load_frag(&Hs[(mi * 16) * 264 + kc * 32], 264);
      v16bf bcur = load_frag(&Bs[(ni * 64) * 264 + kc * 32], 264);
#pragma unroll
      for (int nt = 0; nt < 4; ++nt) {
        v16bf bnxt;
        if (nt < 3) bnxt = load_frag(&Bs[(ni * 64 + (nt + 1) * 16) * 264 + kc * 32], 264);
        macc[nt] = wmma_bf16(a, bcur, macc[nt]);
        if (nt < 3) bcur = bnxt;
      }
    }
    __syncthreads();
  }
  // epilogue: z = xpost + on + moe (+ prob-weighted e_b2); stage, transpose out
#pragma unroll
  for (int nt = 0; nt < 4; ++nt)
#pragma unroll
    for (int r = 0; r < 8; ++r) {
      int row = mi * 16 + r + ((lane >> 4) << 3);
      int col = ni * 64 + nt * 16 + (lane & 15);
      size_t gi = (size_t)(p0 + row) * 128 + col;
      float b2s = Pr[row * 4 + 0] * e_b2[col] + Pr[row * 4 + 1] * e_b2[128 + col] +
                  Pr[row * 4 + 2] * e_b2[256 + col] + Pr[row * 4 + 3] * e_b2[384 + col];
      Zs[row * 133 + col] = macc[nt][r] + b2s + onf[gi] + xpost[gi];
    }
  __syncthreads();
  int pi = tid & 63, cg = tid >> 6, plb = p0 & 4095;
  for (int c = cg; c < 128; c += 4)
    out[((size_t)(n * 128 + c)) * 4096 + plb + pi] = Zs[pi * 133 + c];
}

// ---------------------------------------------------------------------------
extern "C" void kernel_launch(void* const* d_in, const int* in_sizes, int n_in,
                              void* d_out, int out_size, void* d_ws, size_t ws_size,
                              hipStream_t stream) {
  const float* x_re   = (const float*)d_in[0];
  const float* x_im   = (const float*)d_in[1];
  const float* h_re   = (const float*)d_in[2];
  const float* h_im   = (const float*)d_in[3];
  const float* dt     = (const float*)d_in[4];
  const float* flux_re= (const float*)d_in[5];
  const float* flux_im= (const float*)d_in[6];
  const float* ns_g   = (const float*)d_in[7];
  const float* ns_b   = (const float*)d_in[8];
  const float* conv_w = (const float*)d_in[9];
  const float* conv_b = (const float*)d_in[10];
  const float* E_re   = (const float*)d_in[11];
  const float* E_im   = (const float*)d_in[12];
  const float* F_re   = (const float*)d_in[13];
  const float* F_im   = (const float*)d_in[14];
  const float* mix_w  = (const float*)d_in[15];
  const float* mix_b  = (const float*)d_in[16];
  const float* nu     = (const float*)d_in[17];
  const float* theta  = (const float*)d_in[18];
  const float* src_w  = (const float*)d_in[19];
  const float* src_b  = (const float*)d_in[20];
  const float* gate_w = (const float*)d_in[21];
  const float* gate_b = (const float*)d_in[22];
  const float* lam_a  = (const float*)d_in[23];
  const float* lam_b  = (const float*)d_in[24];
  const float* nt_g   = (const float*)d_in[25];
  const float* nt_b   = (const float*)d_in[26];
  const float* r_w    = (const float*)d_in[27];
  const float* r_b    = (const float*)d_in[28];
  const float* e_w1   = (const float*)d_in[29];
  const float* e_b1   = (const float*)d_in[30];
  const float* e_w2   = (const float*)d_in[31];
  const float* e_b2   = (const float*)d_in[32];
  float* out = (float*)d_out;

  char* ws = (char*)d_ws;
  size_t off = 0;
  auto take = [&](size_t bytes) -> char* {
    char* p = ws + off;
    off += (bytes + 255) & ~(size_t)255;
    return p;
  };
  bf16*  buf0 = (bf16*)take((size_t)kNP * 128 * 2);   // xn -> on_bf16
  float* buf1 = (float*)take((size_t)kNP * 128 * 4);  // xflat -> u -> on_f32
  float* buf2 = (float*)take((size_t)kNP * 128 * 4);  // xpost (kept to end)
  float* buf3 = (float*)take((size_t)kNP * 128 * 4);  // x_eigen -> x_dec
  bf16*  wtap = (bf16*)take(9 * 16384 * 2);
  bf16*  wE   = (bf16*)take(16384 * 2);
  bf16*  wF   = (bf16*)take(16384 * 2);
  bf16*  w1b  = (bf16*)take(4 * 32768 * 2);
  bf16*  w2b  = (bf16*)take(4 * 32768 * 2);
  float* xmean  = (float*)take(4096 * 4);
  float* fcat   = (float*)take(4096 * 4);
  float* fluxsq = (float*)take(4096 * 4);
  float* source = (float*)take(4096 * 4);
  float* gate   = (float*)take(2048 * 4);
  float* opdec  = (float*)take(4096 * 4);
  float* opf    = (float*)take(4096 * 4);

  const size_t gemmLds = 2 * 128 * 136 * 2;                       // 69632
  const size_t moeLds  = 64 * 136 * 2 + 64 * 264 * 2 + 69632 +
                         64 * 4 * 4 + 64 * 133 * 4;               // 155904

  k_prep<<<576, 256, 0, stream>>>(conv_w, E_re, E_im, F_re, F_im, e_w1, e_w2,
                                  wtap, wE, wF, w1b, w2b);
  k_ln1<<<kNP / 64, 256, 0, stream>>>(x_re, x_im, ns_g, ns_b, buf0, buf1);
  k_conv<<<kNP / 128, 256, gemmLds, stream>>>(buf0, wtap, conv_b, buf1, buf2);
  k_gemm128<<<kNP / 128, 256, gemmLds, stream>>>(buf2, wE, buf3);
  k_mean<<<kN, 256, 0, stream>>>(buf3, xmean);
  k_mix<<<16, 256, 0, stream>>>(xmean, mix_w, mix_b, fcat);
  k_fluxscan<<<1, 128, 0, stream>>>(fcat, nu, theta, flux_re, flux_im, fluxsq);
  k_srcgate<<<16, 256, 0, stream>>>(fluxsq, src_w, src_b, gate_w, gate_b,
                                    lam_a, lam_b, dt, source, gate, opdec, opf);
  k_forcing<<<kNP * 64 / 256, 256, 0, stream>>>(buf3, gate, source, opf, opdec,
                                                h_re, h_im, buf1);
  k_scanT<<<kB * kP * 64 / 256, 256, 0, stream>>>(opdec, buf1);
  k_gemm128<<<kNP / 128, 256, gemmLds, stream>>>(buf1, wF, buf3);
  k_ln2<<<kNP / 8, 256, 0, stream>>>(buf3, nt_g, nt_b, buf1, buf0);
  k_moe<<<kNP / 64, 256, moeLds, stream>>>(buf0, buf1, buf2, w1b, w2b,
                                           r_w, r_b, e_b1, e_b2, out);
}
